// UnbalancedOT_56573309223309
// MI455X (gfx1250) — matrix operand order repaired and checked by reading, not compiled, not run on hardware
//
#include <hip/hip_runtime.h>
#include <math.h>

#define N_PTS 4096
#define M_PTS 4096
#define DIM   128
#define EPSILON 0.1f
#define TAU 1.0f
#define MAX_ITER 100
#define ROWS_PER_BLK 16

typedef __attribute__((ext_vector_type(16))) _Float16 v16h;
typedef __attribute__((ext_vector_type(8)))  float    v8f;

// ---------------- row norms + init a=b=1 ----------------
__global__ void norms_init_kernel(const float* __restrict__ x,
                                  const float* __restrict__ y,
                                  float* __restrict__ x2,
                                  float* __restrict__ y2,
                                  float* __restrict__ a,
                                  float* __restrict__ b) {
  __shared__ float s[128];
  int row = blockIdx.x;
  int t = threadIdx.x;
  const float* src = (row < N_PTS) ? (x + (size_t)row * DIM)
                                   : (y + (size_t)(row - N_PTS) * DIM);
  float v = src[t];
  s[t] = v * v;
  __syncthreads();
  for (int off = 64; off > 0; off >>= 1) {
    if (t < off) s[t] += s[t + off];
    __syncthreads();
  }
  if (t == 0) {
    if (row < N_PTS) { x2[row] = s[0]; a[row] = 1.0f; }
    else             { y2[row - N_PTS] = s[0]; b[row - N_PTS] = 1.0f; }
  }
}

// ---------------- per-wave 16x16 tile of x @ y^T via WMMA ----------------
__device__ inline v8f tile_xy_wmma(const float* __restrict__ x,
                                   const float* __restrict__ y,
                                   int i0, int j0, int lane) {
  int g  = lane >> 4;   // half-wave group
  int mr = lane & 15;
  v8f c = {};
#pragma unroll
  for (int kb = 0; kb < DIM; kb += 32) {
    v16h A, B;
    // A: row (i0+mr), halves 0..7 -> k = kb+8g+0..7 ; halves 8..15 -> k = kb+16+8g+0..7
    const float4* xr = reinterpret_cast<const float4*>(
        x + (size_t)(i0 + mr) * DIM + kb + 8 * g);
    float4 a0 = xr[0], a1 = xr[1], a2 = xr[4], a3 = xr[5];
    A[0]=(_Float16)a0.x;  A[1]=(_Float16)a0.y;  A[2]=(_Float16)a0.z;  A[3]=(_Float16)a0.w;
    A[4]=(_Float16)a1.x;  A[5]=(_Float16)a1.y;  A[6]=(_Float16)a1.z;  A[7]=(_Float16)a1.w;
    A[8]=(_Float16)a2.x;  A[9]=(_Float16)a2.y;  A[10]=(_Float16)a2.z; A[11]=(_Float16)a2.w;
    A[12]=(_Float16)a3.x; A[13]=(_Float16)a3.y; A[14]=(_Float16)a3.z; A[15]=(_Float16)a3.w;
    // B: column (j0+mr), halves 0..7 -> k = kb+8g+0..7 ; halves 8..15 -> k = kb+16+8g+0..7
    const float4* yr = reinterpret_cast<const float4*>(
        y + (size_t)(j0 + mr) * DIM + kb + 8 * g);
    float4 b0 = yr[0], b1 = yr[1], b2 = yr[4], b3 = yr[5];
    B[0]=(_Float16)b0.x;  B[1]=(_Float16)b0.y;  B[2]=(_Float16)b0.z;  B[3]=(_Float16)b0.w;
    B[4]=(_Float16)b1.x;  B[5]=(_Float16)b1.y;  B[6]=(_Float16)b1.z;  B[7]=(_Float16)b1.w;
    B[8]=(_Float16)b2.x;  B[9]=(_Float16)b2.y;  B[10]=(_Float16)b2.z; B[11]=(_Float16)b2.w;
    B[12]=(_Float16)b3.x; B[13]=(_Float16)b3.y; B[14]=(_Float16)b3.z; B[15]=(_Float16)b3.w;
    c = __builtin_amdgcn_wmma_f32_16x16x32_f16(false, A, false, B, (short)0, c, false, false);
  }
  return c;
}

// ---------------- build K and K^T (once; 128MB -> resident in 192MB L2) ----------------
__global__ void build_K_kernel(const float* __restrict__ x, const float* __restrict__ y,
                               const float* __restrict__ x2, const float* __restrict__ y2,
                               float* __restrict__ K, float* __restrict__ KT) {
  int wave = threadIdx.x >> 5;
  int lane = threadIdx.x & 31;
  int tile = blockIdx.x * 8 + wave;
  int i0 = (tile >> 8) * 16;      // M_PTS/16 == 256 tiles per row
  int j0 = (tile & 255) * 16;
  v8f c = tile_xy_wmma(x, y, i0, j0, lane);
  int g = lane >> 4, mr = lane & 15;
  float y2n = y2[j0 + mr];
  const float scale = -1.0f / EPSILON;
  float kv[8];
#pragma unroll
  for (int r = 0; r < 8; ++r) {
    int m = 8 * g + r;
    float Cd = x2[i0 + m] + y2n - 2.0f * c[r];
    kv[r] = __expf(Cd * scale);
    K[(size_t)(i0 + m) * M_PTS + (j0 + mr)] = kv[r];   // coalesced across lanes
  }
  // KT: each lane owns 8 consecutive floats of row (j0+mr) -> two b128 stores
  float4* ktrow = reinterpret_cast<float4*>(KT + (size_t)(j0 + mr) * N_PTS + i0 + 8 * g);
  ktrow[0] = make_float4(kv[0], kv[1], kv[2], kv[3]);
  ktrow[1] = make_float4(kv[4], kv[5], kv[6], kv[7]);
}

// ------- fused matvec + Sinkhorn update: 16 rows/block, dvec staged to LDS --------
// via gfx1250 async global->LDS copy (ASYNCcnt). Deterministic (shfl reduce).
__global__ void matvec_update_kernel(const float* __restrict__ mat,
                                     const float* __restrict__ dvec,
                                     float* __restrict__ uvec) {
  __shared__ float vbuf[4096];   // 16 KB staged vector
  int t = threadIdx.x;
  int lane = t & 31;
  int wave = t >> 5;

  // async-stage dvec into LDS: 256 threads x 4 x b128 = 16 KB
  unsigned lbase = (unsigned)(uintptr_t)(&vbuf[0]);
#pragma unroll
  for (int k = 0; k < 4; ++k) {
    unsigned laddr = lbase + (unsigned)(t + 256 * k) * 16u;
    unsigned long long ga =
        (unsigned long long)(const void*)(dvec + 4 * (t + 256 * k));
    asm volatile("global_load_async_to_lds_b128 %0, %1, off"
                 :: "v"(laddr), "v"(ga) : "memory");
  }
  asm volatile("s_wait_asynccnt 0x0" ::: "memory");
  __syncthreads();

  int j0 = blockIdx.x * ROWS_PER_BLK;
  int r0 = j0 + wave;      // wave handles rows r0 and r0+8
  int r1 = r0 + 8;
  const float4* row0 = reinterpret_cast<const float4*>(mat + (size_t)r0 * 4096);
  const float4* row1 = reinterpret_cast<const float4*>(mat + (size_t)r1 * 4096);
  const float4* v4   = reinterpret_cast<const float4*>(vbuf);
  float s0 = 0.0f, s1 = 0.0f;
#pragma unroll 8
  for (int k = 0; k < 32; ++k) {
    float4 vv = v4[lane + 32 * k];          // LDS b128, shared by both rows
    float4 m0 = row0[lane + 32 * k];        // 512B contiguous per wave access
    float4 m1 = row1[lane + 32 * k];
    s0 += m0.x * vv.x + m0.y * vv.y + m0.z * vv.z + m0.w * vv.w;
    s1 += m1.x * vv.x + m1.y * vv.y + m1.z * vv.z + m1.w * vv.w;
  }
  for (int off = 16; off > 0; off >>= 1) {
    s0 += __shfl_down(s0, off, 32);
    s1 += __shfl_down(s1, off, 32);
  }
  if (lane == 0) {
    float u0 = uvec[r0];
    float q0 = s0 * u0;
    uvec[r0] = u0 * sqrtf(q0 / (TAU + q0));
    float u1 = uvec[r1];
    float q1 = s1 * u1;
    uvec[r1] = u1 * sqrtf(q1 / (TAU + q1));
  }
}

// ---------------- loss partials: recompute C tiles via WMMA ----------------
__global__ void loss_partial_kernel(const float* __restrict__ x, const float* __restrict__ y,
                                    const float* __restrict__ x2, const float* __restrict__ y2,
                                    const float* __restrict__ a, const float* __restrict__ b,
                                    float* __restrict__ lpart) {
  __shared__ float s[256];
  int wave = threadIdx.x >> 5;
  int lane = threadIdx.x & 31;
  int tile = blockIdx.x * 8 + wave;
  int i0 = (tile >> 8) * 16;
  int j0 = (tile & 255) * 16;
  v8f c = tile_xy_wmma(x, y, i0, j0, lane);
  int g = lane >> 4, mr = lane & 15;
  float bn = b[j0 + mr];
  float y2n = y2[j0 + mr];
  const float scale = -1.0f / EPSILON;
  float acc = 0.0f;
#pragma unroll
  for (int r = 0; r < 8; ++r) {
    int m = 8 * g + r;
    float Cd = x2[i0 + m] + y2n - 2.0f * c[r];
    float Kv = __expf(Cd * scale);   // identical expression/inputs as build -> same bits
    acc += a[i0 + m] * Kv * Cd * bn;
  }
  s[threadIdx.x] = acc;
  __syncthreads();
  for (int off = 128; off > 0; off >>= 1) {
    if (threadIdx.x < off) s[threadIdx.x] += s[threadIdx.x + off];
    __syncthreads();
  }
  if (threadIdx.x == 0) lpart[blockIdx.x] = s[0];
}

__global__ void loss_final_kernel(const float* __restrict__ lpart,
                                  float* __restrict__ out, int n) {
  __shared__ float s[256];
  int t = threadIdx.x;
  float sum = 0.0f;
  for (int i = t; i < n; i += 256) sum += lpart[i];
  s[t] = sum;
  __syncthreads();
  for (int off = 128; off > 0; off >>= 1) {
    if (t < off) s[t] += s[t + off];
    __syncthreads();
  }
  if (t == 0) out[0] = s[0];
}

extern "C" void kernel_launch(void* const* d_in, const int* in_sizes, int n_in,
                              void* d_out, int out_size, void* d_ws, size_t ws_size,
                              hipStream_t stream) {
  (void)in_sizes; (void)n_in; (void)out_size; (void)ws_size;
  const float* x = (const float*)d_in[0];   // source [4096,128] f32
  const float* y = (const float*)d_in[1];   // target [4096,128] f32
  float* ws = (float*)d_ws;
  const size_t NM = (size_t)N_PTS * M_PTS;
  float* K     = ws;                 // 64 MB
  float* KT    = ws + NM;            // 64 MB
  float* x2    = ws + 2 * NM;
  float* y2    = x2 + N_PTS;
  float* a     = y2 + M_PTS;
  float* b     = a + N_PTS;
  float* lpart = b + M_PTS;          // 8192 floats

  const int ntile_blocks = (N_PTS / 16) * (M_PTS / 16) / 8;  // 8192 blocks, 8 waves/blk
  const int mv_blocks_b  = M_PTS / ROWS_PER_BLK;             // 256
  const int mv_blocks_a  = N_PTS / ROWS_PER_BLK;             // 256

  norms_init_kernel<<<N_PTS + M_PTS, 128, 0, stream>>>(x, y, x2, y2, a, b);
  build_K_kernel<<<ntile_blocks, 256, 0, stream>>>(x, y, x2, y2, K, KT);
  for (int it = 0; it < MAX_ITER; ++it) {
    matvec_update_kernel<<<mv_blocks_b, 256, 0, stream>>>(KT, a, b);  // b update (a^T K)
    matvec_update_kernel<<<mv_blocks_a, 256, 0, stream>>>(K,  b, a);  // a update (K b)
  }
  loss_partial_kernel<<<ntile_blocks, 256, 0, stream>>>(x, y, x2, y2, a, b, lpart);
  loss_final_kernel<<<1, 256, 0, stream>>>(lpart, (float*)d_out, ntile_blocks);
}
